// GraphConv_layer_32469952757826
// MI455X (gfx1250) — matrix operand order repaired and checked by reading, not compiled, not run on hardware
//
#include <hip/hip_runtime.h>

// ---------------------------------------------------------------------------
// GraphConv + residual + per-node LayerNorm + ReLU for MI455X (gfx1250).
//   N=100000 nodes, D=128 feats, E=625000 edges.
// Phase 1: zero accumulators (ssum aliased onto d_out, cnt in ws)
// Phase 2: convert W_rel/W_root to bf16 (64KB, stays in L2)
// Phase 3: edge scatter: 1 wave32 per edge, float4/lane, native f32 atomics
// Phase 4: fused per-16-node-tile kernel:
//          agg@W_rel^T + node@W_root^T via v_wmma_f32_16x16x32_bf16 (f32 acc)
//          + bias + residual + LayerNorm + ReLU
// ---------------------------------------------------------------------------

typedef __attribute__((ext_vector_type(16))) __bf16 v16bf;
typedef __attribute__((ext_vector_type(8)))  __bf16 v8bf;
typedef __attribute__((ext_vector_type(8)))  float  v8f;

#define DFEAT 128

__device__ __forceinline__ __bf16 f32_to_bf16(float f) {
  unsigned u = __float_as_uint(f);
  u += 0x7FFFu + ((u >> 16) & 1u);          // round-to-nearest-even
  unsigned short h = (unsigned short)(u >> 16);
  __bf16 r;
  __builtin_memcpy(&r, &h, sizeof(r));
  return r;
}

__global__ void zero_f32_kernel(float* __restrict__ p, size_t n) {
  size_t i = (size_t)blockIdx.x * blockDim.x + threadIdx.x;
  size_t stride = (size_t)gridDim.x * blockDim.x;
  for (; i < n; i += stride) p[i] = 0.0f;
}

__global__ void cvt_weights_kernel(const float* __restrict__ wr,
                                   const float* __restrict__ wo,
                                   __bf16* __restrict__ wrb,
                                   __bf16* __restrict__ wob, int n) {
  int i = blockIdx.x * blockDim.x + threadIdx.x;
  if (i < n) {
    wrb[i] = f32_to_bf16(wr[i]);
    wob[i] = f32_to_bf16(wo[i]);
  }
}

// One wave32 per edge; lane l owns 4 consecutive floats of the 128-feature row.
__global__ void __launch_bounds__(256) scatter_mean_kernel(
    const float* __restrict__ node, const int* __restrict__ eidx,
    const float* __restrict__ eattr, float* ssum, float* cnt, int E) {
  int wid = (int)((blockIdx.x * blockDim.x + threadIdx.x) >> 5);
  int lane = threadIdx.x & 31;
  if (wid >= E) return;                       // wave-uniform
  int src = eidx[wid];
  int dst = eidx[E + wid];
  float w = eattr[wid];
  const float4 v = *(const float4*)(node + (size_t)src * DFEAT + lane * 4);
  float* base = ssum + (size_t)dst * DFEAT + lane * 4;
  unsafeAtomicAdd(base + 0, v.x * w);
  unsafeAtomicAdd(base + 1, v.y * w);
  unsafeAtomicAdd(base + 2, v.z * w);
  unsafeAtomicAdd(base + 3, v.w * w);
  if (lane == 0) unsafeAtomicAdd(cnt + dst, 1.0f);
}

// A fragment (16x32 bf16): lane row m=lane&15; lanes 0-15 hold K {0..7,16..23},
// lanes 16-31 hold K {8..15,24..31}  ->  k0 = kb*32 + (lane>=16 ? 8 : 0)
__device__ __forceinline__ v16bf load_a_frag(const __bf16* row, int k0) {
  v8bf x0 = *(const v8bf*)(row + k0);
  v8bf x1 = *(const v8bf*)(row + k0 + 16);
  v16bf a;
#pragma unroll
  for (int i = 0; i < 8; ++i) { a[i] = x0[i]; a[i + 8] = x1[i]; }
  return a;
}

// B fragment (32x16 bf16): lane col n=lane&15; lanes 0-15 hold K {0..7,16..23},
// lanes 16-31 hold K {8..15,24..31} -> p = W row j + kb*32 + (lane>=16 ? 8 : 0).
__device__ __forceinline__ v16bf load_b_frag(const __bf16* p) {
  v8bf x0 = *(const v8bf*)(p);
  v8bf x1 = *(const v8bf*)(p + 16);
  v16bf b;
#pragma unroll
  for (int i = 0; i < 8; ++i) { b[i] = x0[i]; b[i + 8] = x1[i]; }
  return b;
}

__global__ void __launch_bounds__(256) gconv_fused_kernel(
    const float* __restrict__ node, const float* ssum,
    const float* __restrict__ cnt,
    const __bf16* __restrict__ wrel_bf, const __bf16* __restrict__ wroot_bf,
    const float* __restrict__ b_rel, const float* __restrict__ gamma,
    const float* __restrict__ beta, float* out, int Nn) {
  __shared__ float  s_node[16][DFEAT];
  __shared__ __bf16 s_node_bf[16][DFEAT];
  __shared__ __bf16 s_agg_bf[16][DFEAT];
  __shared__ float  s_out[16][DFEAT];
  __shared__ float  s_mu[16];
  __shared__ float  s_rstd[16];

  const int tid = threadIdx.x;
  const int m0  = blockIdx.x * 16;

  // ---- stage 16x128 tile of node (f32+bf16) and agg=ssum/max(cnt,1) (bf16)
  {
    int r = tid >> 4;            // 16 threads per row
    int c = (tid & 15) * 8;      // 8 consecutive floats per thread
    int grow = m0 + r;
    float nv[8] = {0, 0, 0, 0, 0, 0, 0, 0};
    float sv[8] = {0, 0, 0, 0, 0, 0, 0, 0};
    float inv = 1.0f;
    if (grow < Nn) {
      inv = 1.0f / fmaxf(cnt[grow], 1.0f);
      const float4* np = (const float4*)(node + (size_t)grow * DFEAT + c);
      const float4* sp = (const float4*)(ssum + (size_t)grow * DFEAT + c);
      float4 n0 = np[0], n1 = np[1], t0 = sp[0], t1 = sp[1];
      nv[0] = n0.x; nv[1] = n0.y; nv[2] = n0.z; nv[3] = n0.w;
      nv[4] = n1.x; nv[5] = n1.y; nv[6] = n1.z; nv[7] = n1.w;
      sv[0] = t0.x; sv[1] = t0.y; sv[2] = t0.z; sv[3] = t0.w;
      sv[4] = t1.x; sv[5] = t1.y; sv[6] = t1.z; sv[7] = t1.w;
    }
#pragma unroll
    for (int i = 0; i < 8; ++i) {
      s_node[r][c + i]    = nv[i];
      s_node_bf[r][c + i] = f32_to_bf16(nv[i]);
      s_agg_bf[r][c + i]  = f32_to_bf16(sv[i] * inv);
    }
  }
  __syncthreads();

  // ---- per-wave 16x16 WMMA output tile; wave w covers features [16w,16w+16)
  const int wave = tid >> 5;
  const int lane = tid & 31;
  const int hsel = lane >> 4;        // 0: lanes 0-15, 1: lanes 16-31
  const int l15  = lane & 15;
  const int j    = wave * 16 + l15;  // output feature owned by this lane

  v8f acc = {0.f, 0.f, 0.f, 0.f, 0.f, 0.f, 0.f, 0.f};
#pragma unroll
  for (int kb = 0; kb < 4; ++kb) {   // agg @ W_rel^T
    v16bf a = load_a_frag(&s_agg_bf[l15][0], kb * 32 + hsel * 8);
    v16bf b = load_b_frag(wrel_bf + (size_t)j * DFEAT + kb * 32 + hsel * 8);
    acc = __builtin_amdgcn_wmma_f32_16x16x32_bf16(false, a, false, b,
                                                  (short)0, acc, false, false);
  }
#pragma unroll
  for (int kb = 0; kb < 4; ++kb) {   // node @ W_root^T
    v16bf a = load_a_frag(&s_node_bf[l15][0], kb * 32 + hsel * 8);
    v16bf b = load_b_frag(wroot_bf + (size_t)j * DFEAT + kb * 32 + hsel * 8);
    acc = __builtin_amdgcn_wmma_f32_16x16x32_bf16(false, a, false, b,
                                                  (short)0, acc, false, false);
  }

  // C/D layout: VGPR v, lanes 0-15 -> M=v, lanes 16-31 -> M=v+8, N=lane&15
  const float bj = b_rel[j];
#pragma unroll
  for (int v = 0; v < 8; ++v) {
    int m = v + hsel * 8;
    s_out[m][j] = s_node[m][j] + acc[v] + bj;   // residual + bias + GEMMs
  }
  __syncthreads();

  // ---- per-row LayerNorm stats: wave w reduces rows w and w+8 (wave32 shfl)
#pragma unroll
  for (int rr = 0; rr < 2; ++rr) {
    int row = wave + rr * 8;
    float4 v = *(const float4*)(&s_out[row][0] + lane * 4);
    float sum = v.x + v.y + v.z + v.w;
    float sq  = v.x * v.x + v.y * v.y + v.z * v.z + v.w * v.w;
#pragma unroll
    for (int off = 16; off >= 1; off >>= 1) {
      sum += __shfl_xor(sum, off);
      sq  += __shfl_xor(sq, off);
    }
    if (lane == 0) {
      float mu  = sum * (1.0f / DFEAT);
      float var = sq * (1.0f / DFEAT) - mu * mu;
      s_mu[row]   = mu;
      s_rstd[row] = rsqrtf(var + 1e-5f);
    }
  }
  __syncthreads();

  // ---- normalize + affine + ReLU + store
  {
    int r = tid >> 4;
    int c = (tid & 15) * 8;
    int grow = m0 + r;
    if (grow < Nn) {
      float mu = s_mu[r], rs = s_rstd[r];
      float o[8];
#pragma unroll
      for (int i = 0; i < 8; ++i) {
        float y = (s_out[r][c + i] - mu) * rs * gamma[c + i] + beta[c + i];
        o[i] = fmaxf(y, 0.0f);
      }
      float4* op = (float4*)(out + (size_t)grow * DFEAT + c);
      op[0] = make_float4(o[0], o[1], o[2], o[3]);
      op[1] = make_float4(o[4], o[5], o[6], o[7]);
    }
  }
}

extern "C" void kernel_launch(void* const* d_in, const int* in_sizes, int n_in,
                              void* d_out, int out_size, void* d_ws,
                              size_t ws_size, hipStream_t stream) {
  const float* node   = (const float*)d_in[0];
  const int*   eidx   = (const int*)d_in[1];   // [2,E]: row0=src, row1=dst
  const float* eattr  = (const float*)d_in[2];
  // d_in[3] = batch_ptr (unused by reference math)
  const float* W_rel  = (const float*)d_in[4];
  const float* b_rel  = (const float*)d_in[5];
  const float* W_root = (const float*)d_in[6];
  const float* gamma  = (const float*)d_in[7];
  const float* beta   = (const float*)d_in[8];

  const int Dd = in_sizes[5];                  // 128
  const int Nn = in_sizes[0] / Dd;             // 100000
  const int E  = in_sizes[2];                  // 625000

  float* out  = (float*)d_out;
  float* ssum = out;                           // alias: accumulate in d_out,
                                               // fused kernel consumes rows
                                               // before overwriting them
  float*  cnt    = (float*)d_ws;
  size_t  off    = (((size_t)Nn * sizeof(float)) + 255) & ~(size_t)255;
  __bf16* wrelb  = (__bf16*)((char*)d_ws + off);
  __bf16* wrootb = wrelb + (size_t)Dd * Dd;

  zero_f32_kernel<<<2048, 256, 0, stream>>>(out, (size_t)Nn * Dd);
  zero_f32_kernel<<<512, 256, 0, stream>>>(cnt, (size_t)Nn);
  cvt_weights_kernel<<<(Dd * Dd + 255) / 256, 256, 0, stream>>>(
      W_rel, W_root, wrelb, wrootb, Dd * Dd);

  int scatter_blocks = (E * 32 + 255) / 256;   // one wave32 per edge
  scatter_mean_kernel<<<scatter_blocks, 256, 0, stream>>>(node, eidx, eattr,
                                                          ssum, cnt, E);

  gconv_fused_kernel<<<(Nn + 15) / 16, 256, 0, stream>>>(
      node, ssum, cnt, wrelb, wrootb, b_rel, gamma, beta, out, Nn);
}
